// TppBallNet_59115929862968
// MI455X (gfx1250) — matrix operand order repaired and checked by reading, not compiled, not run on hardware
//
#include <hip/hip_runtime.h>
#include <hip/hip_bf16.h>
#include <math.h>

#define NPTS   1000
#define KNN    8
#define NPAIRS ((NPTS * (NPTS - 1)) / 2)   // 499500

typedef __attribute__((ext_vector_type(16))) __bf16 v16bf;
typedef __attribute__((ext_vector_type(8)))  float  v8f;

union U16V { v16bf v; unsigned short u[16]; uint4 q[2]; };
union U8FV { v8f   v; float          f[8];  };

__device__ __forceinline__ unsigned short f2bf(float f) {
  unsigned int x = __float_as_uint(f);
  unsigned int r = x + 0x7FFFu + ((x >> 16) & 1u);   // round-to-nearest-even
  return (unsigned short)(r >> 16);
}

// ---------------- zero fill (ushort) ----------------
__global__ void k_zero16(unsigned short* __restrict__ p, int n) {
  int i = blockIdx.x * blockDim.x + threadIdx.x;
  if (i < n) p[i] = 0;
}

// ------- weight conversion: f32 K x N  ->  bf16 transposed N x Kpad (zero pad) -------
__global__ void k_cvt_t(const float* __restrict__ src, unsigned short* __restrict__ dst,
                        int K, int N, int Kpad) {
  int tid = blockIdx.x * blockDim.x + threadIdx.x;
  if (tid >= N * Kpad) return;
  int n = tid / Kpad, k = tid % Kpad;
  dst[tid] = (k < K) ? f2bf(src[(size_t)k * N + n]) : (unsigned short)0;
}

// ---------------- kNN (one wave32 per query point) ----------------
__global__ void k_knn(const float* __restrict__ X, int D, int* __restrict__ nbr) {
  int gt   = blockIdx.x * blockDim.x + threadIdx.x;
  int wave = gt >> 5;
  int lane = gt & 31;
  if (wave >= NPTS) return;
  const float* xi = X + (size_t)wave * D;

  float bd[KNN]; int bi[KNN];
#pragma unroll
  for (int t = 0; t < KNN; ++t) { bd[t] = 3.0e38f; bi[t] = 0; }

  for (int j = lane; j < NPTS; j += 32) {
    const float* xj = X + (size_t)j * D;
    float d2 = 0.f;
    for (int d = 0; d < D; ++d) { float t = xi[d] - xj[d]; d2 = fmaf(t, t, d2); }
    if (d2 < bd[KNN - 1]) {
      bd[KNN - 1] = d2; bi[KNN - 1] = j;
#pragma unroll
      for (int t = KNN - 1; t > 0; --t) {
        if (bd[t] < bd[t - 1]) {
          float td = bd[t]; bd[t] = bd[t - 1]; bd[t - 1] = td;
          int   ti = bi[t]; bi[t] = bi[t - 1]; bi[t - 1] = ti;
        }
      }
    }
  }
#pragma unroll
  for (int t = 0; t < KNN; ++t) {
    unsigned long long key =
        (((unsigned long long)__float_as_uint(bd[0])) << 32) | (unsigned int)bi[0];
    unsigned long long mn = key;
#pragma unroll
    for (int off = 16; off > 0; off >>= 1) {
      unsigned long long o = __shfl_xor(mn, off, 32);
      if (o < mn) mn = o;
    }
    unsigned long long ball = __ballot(key == mn);
    int winner = (int)(__ffsll(ball) - 1);
    if (lane == 0) nbr[wave * KNN + t] = (int)(mn & 0xffffffffu);
    if (lane == winner) {
#pragma unroll
      for (int s = 0; s < KNN - 1; ++s) { bd[s] = bd[s + 1]; bi[s] = bi[s + 1]; }
      bd[KNN - 1] = 3.0e38f; bi[KNN - 1] = 0;
    }
  }
}

// ------- edge gather: row stride Wpad, cols [xi | xj-xi | zero pad] -> bf16 -------
__global__ void k_edges(const float* __restrict__ X, const int* __restrict__ nbr,
                        unsigned short* __restrict__ E, int D, int Wpad) {
  int tid = blockIdx.x * blockDim.x + threadIdx.x;
  if (tid >= NPTS * KNN * Wpad) return;
  int col = tid % Wpad;
  int r   = tid / Wpad;
  int i   = r / KNN;
  unsigned short v = 0;
  if (col < D) {
    v = f2bf(X[(size_t)i * D + col]);
  } else if (col < 2 * D) {
    int d = col - D;
    int j = nbr[r];
    v = f2bf(X[(size_t)j * D + d] - X[(size_t)i * D + d]);
  }
  E[(size_t)r * Wpad + col] = v;
}

// ------------- bf16 WMMA GEMM: C = relu?(A[MxKpad] @ Bt[NxKpad]^T + bias) -------------
// A: row-major, stride Kpad (mult of 32, zero-padded cols). Bt: weights transposed,
// row n holds column n of W, stride Kpad. One wave computes TWO 16x16 C tiles stacked
// in M, sharing the B fragment: per 32-wide k-step -> 6x global_load_b128 + 2x WMMA.
__global__ __launch_bounds__(128)
void k_gemm(const unsigned short* __restrict__ A, const unsigned short* __restrict__ Bt,
            const float* __restrict__ bias, float* __restrict__ Cf,
            unsigned short* __restrict__ Cb, int M, int Kpad, int N, int ldcb, int relu) {
  int wave = (blockIdx.x * blockDim.x + threadIdx.x) >> 5;
  int lane = threadIdx.x & 31;
  int mpairs = (M + 31) >> 5;                    // pairs of 16-row tiles
  int ntiles = N >> 4;
  if (wave >= mpairs * ntiles) return;           // wave-uniform: EXEC stays all-1s
  int mtp = wave / ntiles, nt = wave % ntiles;
  int half = lane >> 4;
  int m0 = mtp * 32 + (lane & 15);               // A row, tile 0 (alloc covers +32)
  int n  = nt * 16 + (lane & 15);                // B/C col for this lane

  const uint4* Arow0 = (const uint4*)(A  + (size_t)m0 * Kpad);        // 8 bf16/uint4
  const uint4* Arow1 = (const uint4*)(A  + (size_t)(m0 + 16) * Kpad);
  const uint4* Brow  = (const uint4*)(Bt + (size_t)n * Kpad);
  v8f acc0 = {}, acc1 = {};
  int k8n = Kpad >> 3;                           // #uint4 per row
  for (int k8 = 0; k8 < k8n; k8 += 4) {          // 32 k per step
    U16V a0, a1, b;
    b.q[0]  = Brow[k8 + half];                   // k = k0 + half*8 .. +7
    b.q[1]  = Brow[k8 + 2 + half];               // k = k0 + 16 + half*8 .. +7
    a0.q[0] = Arow0[k8 + half];                  // k = k0 + half*8 .. +7
    a0.q[1] = Arow0[k8 + 2 + half];              // k = k0 + 16 + half*8 .. +7
    a1.q[0] = Arow1[k8 + half];
    a1.q[1] = Arow1[k8 + 2 + half];
    acc0 = __builtin_amdgcn_wmma_f32_16x16x32_bf16(false, a0.v, false, b.v,
                                                   (short)0, acc0, false, false);
    acc1 = __builtin_amdgcn_wmma_f32_16x16x32_bf16(false, a1.v, false, b.v,
                                                   (short)0, acc1, false, false);
  }
  float bv = bias ? bias[n] : 0.f;
  U8FV o0, o1; o0.v = acc0; o1.v = acc1;
#pragma unroll
  for (int v = 0; v < 8; ++v) {                  // C/D: row = v + 8*half, col = lane%16
    int r0 = mtp * 32 + v + 8 * half;
    int r1 = r0 + 16;
    if (r0 < M) {
      float val = o0.f[v] + bv;
      if (relu && val < 0.f) val = 0.f;
      if (Cf) Cf[(size_t)r0 * N + n] = val;
      if (Cb) Cb[(size_t)r0 * ldcb + n] = f2bf(val);
    }
    if (r1 < M) {
      float val = o1.f[v] + bv;
      if (relu && val < 0.f) val = 0.f;
      if (Cf) Cf[(size_t)r1 * N + n] = val;
      if (Cb) Cb[(size_t)r1 * ldcb + n] = f2bf(val);
    }
  }
}

// ------------- max over the 8 neighbor rows of each point -------------
__global__ void k_rowmax8(const float* __restrict__ H, float* __restrict__ X, int N) {
  int tid = blockIdx.x * blockDim.x + threadIdx.x;
  if (tid >= NPTS * N) return;
  int i = tid / N, n = tid % N;
  float m = H[(size_t)(i * KNN) * N + n];
#pragma unroll
  for (int e = 1; e < KNN; ++e) m = fmaxf(m, H[(size_t)(i * KNN + e) * N + n]);
  X[tid] = m;
}

// ------------- concat x1|x2|x3 -> bf16 (1000 x 672) -------------
__global__ void k_concat(const float* __restrict__ x1, const float* __restrict__ x2,
                         const float* __restrict__ x3, unsigned short* __restrict__ out) {
  int tid = blockIdx.x * blockDim.x + threadIdx.x;
  if (tid >= NPTS * 672) return;
  int row = tid / 672, c = tid % 672;
  float v = (c < 32) ? x1[row * 32 + c]
          : (c < 160) ? x2[row * 128 + (c - 32)]
                      : x3[(size_t)row * 512 + (c - 160)];
  out[tid] = f2bf(v);
}

// ------------- global max-pool over points (128 cols) -------------
__global__ void k_colmax(const float* __restrict__ sf, float* __restrict__ g) {
  int n = blockIdx.x * blockDim.x + threadIdx.x;
  if (n >= 128) return;
  float m = sf[n];
  for (int i = 1; i < NPTS; ++i) m = fmaxf(m, sf[(size_t)i * 128 + n]);
  g[n] = m;
}

// ------------- c = g @ wc1[256:384] + bc1 (f32, 128 wide) -------------
__global__ void k_cvec(const float* __restrict__ g, const float* __restrict__ wc1,
                       const float* __restrict__ bc1, float* __restrict__ c) {
  int n = blockIdx.x * blockDim.x + threadIdx.x;
  if (n >= 128) return;
  float acc = bc1[n];
  for (int d = 0; d < 128; ++d) acc = fmaf(g[d], wc1[(size_t)(256 + d) * 128 + n], acc);
  c[n] = acc;
}

__device__ __forceinline__ long long tri_cum(int i) {
  return (long long)i * (2 * NPTS - 1 - i) / 2;   // pairs before row i
}

// ------------- pair classifier: logit = relu(A_i + B_j + c) . w2 + b -------------
__global__ void k_pairs(const float* __restrict__ Ap, const float* __restrict__ Bp,
                        const float* __restrict__ c,  const float* __restrict__ w2,
                        const float* __restrict__ bc2,
                        float* __restrict__ outS, float* __restrict__ outL) {
  int p = blockIdx.x * blockDim.x + threadIdx.x;
  if (p >= NPAIRS) return;
  double disc = (double)1999 * 1999 - 8.0 * (double)p;
  int i = (int)((1999.0 - sqrt(disc)) * 0.5);
  if (i < 0) i = 0; if (i > NPTS - 2) i = NPTS - 2;
  while (i > 0 && tri_cum(i) > (long long)p) --i;
  while (i < NPTS - 2 && tri_cum(i + 1) <= (long long)p) ++i;
  int j = i + 1 + (int)((long long)p - tri_cum(i));

  const float4* a4 = (const float4*)(Ap + (size_t)i * 128);
  const float4* b4 = (const float4*)(Bp + (size_t)j * 128);
  const float4* c4 = (const float4*)c;
  const float4* w4 = (const float4*)w2;
  float acc = 0.f;
#pragma unroll
  for (int q = 0; q < 32; ++q) {
    float4 a = a4[q], b = b4[q], cc = c4[q], w = w4[q];
    float h;
    h = a.x + b.x + cc.x; if (h > 0.f) acc = fmaf(h, w.x, acc);
    h = a.y + b.y + cc.y; if (h > 0.f) acc = fmaf(h, w.y, acc);
    h = a.z + b.z + cc.z; if (h > 0.f) acc = fmaf(h, w.z, acc);
    h = a.w + b.w + cc.w; if (h > 0.f) acc = fmaf(h, w.w, acc);
  }
  float logit = acc + bc2[0];
  outL[p] = logit;
  outS[p] = 1.0f / (1.0f + expf(-logit));
}

// ---------------------------------------------------------------------------
static inline void gemm(const unsigned short* A, const unsigned short* Bt, const float* bias,
                        float* Cf, unsigned short* Cb, int M, int Kpad, int N, int ldcb,
                        int relu, hipStream_t s) {
  int tiles  = ((M + 31) / 32) * (N / 16);         // 2 M-tiles per wave
  int blocks = (tiles + 3) / 4;                    // 4 waves / 128-thread block
  k_gemm<<<blocks, 128, 0, s>>>(A, Bt, bias, Cf, Cb, M, Kpad, N, ldcb, relu);
}

extern "C" void kernel_launch(void* const* d_in, const int* in_sizes, int n_in,
                              void* d_out, int out_size, void* d_ws, size_t ws_size,
                              hipStream_t stream) {
  (void)n_in; (void)out_size; (void)ws_size;
  const float* pos = (const float*)d_in[0];
  const float* w1a = (const float*)d_in[2];  const float* b1a = (const float*)d_in[3];
  const float* w1b = (const float*)d_in[4];  const float* b1b = (const float*)d_in[5];
  const float* w2a = (const float*)d_in[6];  const float* b2a = (const float*)d_in[7];
  const float* w2b = (const float*)d_in[8];  const float* b2b = (const float*)d_in[9];
  const float* w3a = (const float*)d_in[10]; const float* b3a = (const float*)d_in[11];
  const float* w3b = (const float*)d_in[12]; const float* b3b = (const float*)d_in[13];
  const float* ws1 = (const float*)d_in[14]; const float* bs1 = (const float*)d_in[15];
  const float* ws2 = (const float*)d_in[16]; const float* bs2 = (const float*)d_in[17];
  const float* wc1 = (const float*)d_in[18]; const float* bc1 = (const float*)d_in[19];
  const float* wc2 = (const float*)d_in[20]; const float* bc2 = (const float*)d_in[21];

  int Bn = in_sizes[0] / (NPTS * 3);
  char* base = (char*)d_ws;
  size_t off = 0;
  auto carve = [&](size_t bytes) -> char* {
    char* p = base + off; off += (bytes + 255) & ~(size_t)255; return p;
  };
  const int MP = 1024;                         // NPTS padded to multiple of 32 (2-tile waves)
  // weights, bf16, transposed N x Kpad
  unsigned short* w1a_t = (unsigned short*)carve(16  * 32  * 2);
  unsigned short* w1b_t = (unsigned short*)carve(32  * 32  * 2);
  unsigned short* w2a_t = (unsigned short*)carve(64  * 64  * 2);
  unsigned short* w2b_t = (unsigned short*)carve(128 * 64  * 2);
  unsigned short* w3a_t = (unsigned short*)carve(256 * 256 * 2);
  unsigned short* w3b_t = (unsigned short*)carve(512 * 256 * 2);
  unsigned short* ws1_t = (unsigned short*)carve(256 * 672 * 2);
  unsigned short* ws2_t = (unsigned short*)carve(128 * 256 * 2);
  unsigned short* wc1a_t = (unsigned short*)carve(128 * 128 * 2);
  unsigned short* wc1b_t = (unsigned short*)carve(128 * 128 * 2);
  int*            nbr   = (int*)carve((size_t)NPTS * KNN * 4);
  unsigned short* E     = (unsigned short*)carve((size_t)NPTS * KNN * 256 * 2);
  unsigned short* Hb    = (unsigned short*)carve((size_t)NPTS * KNN * 256 * 2);
  float*          H2    = (float*)carve((size_t)NPTS * KNN * 512 * 4);
  float*          x1    = (float*)carve((size_t)NPTS * 32 * 4);
  float*          x2    = (float*)carve((size_t)NPTS * 128 * 4);
  float*          x3    = (float*)carve((size_t)NPTS * 512 * 4);
  unsigned short* xcat  = (unsigned short*)carve((size_t)MP * 672 * 2);
  float*          sf    = (float*)carve((size_t)NPTS * 128 * 4);
  unsigned short* sfb   = (unsigned short*)carve((size_t)MP * 128 * 2);
  float*          g     = (float*)carve(128 * 4);
  float*          cv    = (float*)carve(128 * 4);
  float*          Ap    = (float*)carve((size_t)NPTS * 128 * 4);
  float*          Bp    = (float*)carve((size_t)NPTS * 128 * 4);

  auto cvt = [&](const float* s, unsigned short* d, int K, int N, int Kpad) {
    int n = N * Kpad;
    k_cvt_t<<<(n + 255) / 256, 256, 0, stream>>>(s, d, K, N, Kpad);
  };
  cvt(w1a, w1a_t, 6, 16, 32);     cvt(w1b, w1b_t, 16, 32, 32);
  cvt(w2a, w2a_t, 64, 64, 64);    cvt(w2b, w2b_t, 64, 128, 64);
  cvt(w3a, w3a_t, 256, 256, 256); cvt(w3b, w3b_t, 256, 512, 256);
  cvt(ws1, ws1_t, 672, 256, 672); cvt(ws2, ws2_t, 256, 128, 256);
  cvt(wc1,             wc1a_t, 128, 128, 128);
  cvt(wc1 + 128 * 128, wc1b_t, 128, 128, 128);

  const int ME = NPTS * KNN;   // 8000 edge rows (multiple of 32)
  for (int b = 0; b < Bn; ++b) {
    const float* pb = pos + (size_t)b * NPTS * 3;
    float* outS = (float*)d_out + (size_t)b * NPAIRS;
    float* outL = (float*)d_out + ((size_t)Bn + b) * NPAIRS;

    // EdgeConv 1 (D=3 -> 32); hidden is 16-wide in a 32-stride buffer: pre-zero pad
    k_knn<<<(NPTS * 32 + 255) / 256, 256, 0, stream>>>(pb, 3, nbr);
    k_edges<<<(ME * 32 + 255) / 256, 256, 0, stream>>>(pb, nbr, E, 3, 32);
    k_zero16<<<(ME * 32 + 255) / 256, 256, 0, stream>>>(Hb, ME * 32);
    gemm(E, w1a_t, b1a, nullptr, Hb, ME, 32, 16, 32, 1, stream);
    gemm(Hb, w1b_t, b1b, H2, nullptr, ME, 32, 32, 0, 0, stream);
    k_rowmax8<<<(NPTS * 32 + 255) / 256, 256, 0, stream>>>(H2, x1, 32);

    // EdgeConv 2 (D=32 -> 128)
    k_knn<<<(NPTS * 32 + 255) / 256, 256, 0, stream>>>(x1, 32, nbr);
    k_edges<<<(ME * 64 + 255) / 256, 256, 0, stream>>>(x1, nbr, E, 32, 64);
    gemm(E, w2a_t, b2a, nullptr, Hb, ME, 64, 64, 64, 1, stream);
    gemm(Hb, w2b_t, b2b, H2, nullptr, ME, 64, 128, 0, 0, stream);
    k_rowmax8<<<(NPTS * 128 + 255) / 256, 256, 0, stream>>>(H2, x2, 128);

    // EdgeConv 3 (D=128 -> 512)
    k_knn<<<(NPTS * 32 + 255) / 256, 256, 0, stream>>>(x2, 128, nbr);
    k_edges<<<(ME * 256 + 255) / 256, 256, 0, stream>>>(x2, nbr, E, 128, 256);
    gemm(E, w3a_t, b3a, nullptr, Hb, ME, 256, 256, 256, 1, stream);
    gemm(Hb, w3b_t, b3b, H2, nullptr, ME, 256, 512, 0, 0, stream);
    k_rowmax8<<<(NPTS * 512 + 255) / 256, 256, 0, stream>>>(H2, x3, 512);

    // Score head: 672 -> 256 -> 128, global max-pool
    k_concat<<<(NPTS * 672 + 255) / 256, 256, 0, stream>>>(x1, x2, x3, xcat);
    gemm(xcat, ws1_t, bs1, nullptr, Hb, NPTS, 672, 256, 256, 1, stream);
    gemm(Hb, ws2_t, bs2, sf, sfb, NPTS, 256, 128, 128, 0, stream);
    k_colmax<<<1, 128, 0, stream>>>(sf, g);

    // Factorized pair head: A = sf@W1, B = sf@W2, c = g@W3 + bc1
    gemm(sfb, wc1a_t, nullptr, Ap, nullptr, NPTS, 128, 128, 0, 0, stream);
    gemm(sfb, wc1b_t, nullptr, Bp, nullptr, NPTS, 128, 128, 0, 0, stream);
    k_cvec<<<1, 128, 0, stream>>>(g, wc1, bc1, cv);
    k_pairs<<<(NPAIRS + 255) / 256, 256, 0, stream>>>(Ap, Bp, cv, wc2, bc2, outS, outL);
  }
}
